// Seg2dLossless_41609643164133
// MI455X (gfx1250) — compile-verified, matrix operand order checked by reading, not run on hardware
//
#include <hip/hip_runtime.h>
#include <hip/hip_bf16.h>

typedef __attribute__((ext_vector_type(16))) _Float16 v16h;
typedef __attribute__((ext_vector_type(8)))  float    v8f;

#define DHID 128

// Branch-free activations: clamped exp2-based tanh/sigmoid lower to
// v_max/v_min/v_exp_f32/v_rcp_f32 with no exec-mask divergence (unlike OCML
// tanhf/expf which emit s_and_saveexec regions).
__device__ __forceinline__ float fast_tanh(float x) {
    float xc = fminf(fmaxf(x, -16.f), 16.f);
    float e  = __builtin_amdgcn_exp2f(xc * 2.88539008177f);   // e^{2x}
    return (e - 1.f) * __builtin_amdgcn_rcpf(e + 1.f);
}
__device__ __forceinline__ float fast_sigmoid(float x) {
    float xc = fminf(fmaxf(x, -30.f), 30.f);
    float e  = __builtin_amdgcn_exp2f(xc * 1.44269504089f);   // e^{x}
    return e * __builtin_amdgcn_rcpf(e + 1.f);                // e^x/(1+e^x)
}

// ---------------------------------------------------------------------------
// Prep: W2 (f32, [K][N] row-major) -> transposed f16 [N][K] so that both A and
// B WMMA fragments read contiguous (K, K+1) pairs from LDS.
// ---------------------------------------------------------------------------
__global__ void w2_to_f16t_kernel(const float* __restrict__ W2,
                                  _Float16* __restrict__ w2t) {
    int idx = blockIdx.x * blockDim.x + threadIdx.x;
    if (idx >= DHID * DHID) return;
    int n = idx >> 7;
    int k = idx & 127;
    w2t[idx] = (_Float16)W2[k * DHID + n];
}

// ---------------------------------------------------------------------------
// MLP grid evaluation with WMMA.
// Block = 256 threads = 8 wave32; each block evaluates 128 grid points.
// Wave w owns M-tile (points 16w..16w+15). A fragments (4 K-blocks) are loaded
// once and held in 32 VGPRs; the wave then sweeps all 8 N-tiles:
// 8 N-tiles x 4 K-steps = 32 v_wmma_f32_16x16x32_f16 per wave.
// ---------------------------------------------------------------------------
__global__ __launch_bounds__(256) void eval_grid_kernel(
    const float* __restrict__ W1, const float* __restrict__ b1,
    const float* __restrict__ b2,
    const float* __restrict__ W3, const float* __restrict__ b3,
    const float* __restrict__ bmin, const float* __restrict__ bmax,
    const _Float16* __restrict__ w2t,   // [N][K] f16 (transposed)
    float* __restrict__ out, int R)
{
    __shared__ _Float16 h1s[128][DHID];   // 32 KB: layer-1 activations (A)
    __shared__ _Float16 w2s[DHID][DHID];  // 32 KB: W2^T               (B)

    const int tid   = threadIdx.x;
    const int lane  = tid & 31;
    const int wave  = tid >> 5;
    const int npts  = R * R;
    const int base  = blockIdx.x * 128;

    // ---- Stage W2^T into LDS (2048 uint4 / 256 threads = 8 each) ----
    {
        const uint4* src = reinterpret_cast<const uint4*>(w2t);
        uint4*       dst = reinterpret_cast<uint4*>(&w2s[0][0]);
        __builtin_prefetch(src + tid, 0, 1);   // global_prefetch_b8
#pragma unroll
        for (int t = 0; t < 8; ++t)
            dst[tid + 256 * t] = src[tid + 256 * t];
    }

    // ---- Layer 1: h1 = tanh(pts @ W1 + b1), written to LDS as f16 ----
    {
        const float inv = 1.0f / (float)(R - 1);
        const float bx = bmin[0], by = bmin[1];
        const float sx = bmax[0] - bmin[0], sy = bmax[1] - bmin[1];
        const int p  = tid >> 1;            // local point 0..127
        const int k0 = (tid & 1) * 64;      // half of the 128 features
        const int idx = base + p;
        float x = 0.f, y = 0.f;
        const bool ok = idx < npts;
        if (ok) {
            int i = idx / R;
            int j = idx - i * R;
            x = bx + (float)j * inv * sx;   // X follows u (columns)
            y = by + (float)i * inv * sy;   // Y follows v (rows)
        }
#pragma unroll 8
        for (int k = k0; k < k0 + 64; ++k) {
            float hv = ok ? fast_tanh(x * W1[k] + y * W1[DHID + k] + b1[k]) : 0.f;
            h1s[p][k] = (_Float16)hv;
        }
    }
    __syncthreads();

    const int half16 = lane >> 4;       // 0: lanes 0-15, 1: lanes 16-31
    const int l15    = lane & 15;
    const int arow   = wave * 16 + l15; // A-matrix row (M) for this lane

    // ---- Preload all 4 A-fragments (K-blocks) for this wave's M-tile ----
    v16h afrag[4];
#pragma unroll
    for (int kb = 0; kb < 4; ++kb) {
#pragma unroll
        for (int v = 0; v < 8; ++v) {
            // 16-bit A 16x32 layout: VGPR v, lane-half -> K offset
            const int K = kb * 32 + ((v & 4) ? 16 : 0)
                        + (half16 ? 8 : 0) + 2 * (v & 3);
            afrag[kb][2 * v]     = h1s[arow][K];
            afrag[kb][2 * v + 1] = h1s[arow][K + 1];
        }
    }

    // ---- Layer 2 (WMMA) fused with layer-3 partial dot product ----
    float acc3[8];
#pragma unroll
    for (int r = 0; r < 8; ++r) acc3[r] = 0.f;

#pragma unroll
    for (int nt = 0; nt < 8; ++nt) {
        v8f c = {};
        const int ncol = nt * 16 + l15; // B-matrix column (N) for this lane
#pragma unroll
        for (int kb = 0; kb < 4; ++kb) {
            v16h b;
#pragma unroll
            for (int v = 0; v < 8; ++v) {
                const int K = kb * 32 + ((v & 4) ? 16 : 0)
                            + (half16 ? 8 : 0) + 2 * (v & 3);
                b[2 * v]     = w2s[ncol][K];
                b[2 * v + 1] = w2s[ncol][K + 1];
            }
            c = __builtin_amdgcn_wmma_f32_16x16x32_f16(
                    false, afrag[kb], false, b, (short)0, c, false, false);
        }
        const float b2n = b2[ncol];
        const float w3n = W3[ncol];
#pragma unroll
        for (int r = 0; r < 8; ++r) {
            float h2 = fast_tanh(c[r] + b2n);
            acc3[r] += h2 * w3n;          // layer-3 partial over this lane's N
        }
    }

    // ---- Reduce layer-3 partials across the 16 lanes of each half ----
#pragma unroll
    for (int r = 0; r < 8; ++r) {
        float s = acc3[r];
        s += __shfl_xor(s, 1, 32);
        s += __shfl_xor(s, 2, 32);
        s += __shfl_xor(s, 4, 32);
        s += __shfl_xor(s, 8, 32);
        acc3[r] = s;
    }
    const float b3v = b3[0];
    if (l15 == 0) {
#pragma unroll
        for (int r = 0; r < 8; ++r) {
            const int M   = r + 8 * half16;   // C/D layout: VGPR r -> row
            const int idx = base + wave * 16 + M;
            if (idx < npts) {
                out[idx] = fast_sigmoid(acc3[r] + b3v);
            }
        }
    }
}

// ---------------------------------------------------------------------------
// Exact 2x-1 bilinear upsample (align_corners=True).
// ---------------------------------------------------------------------------
__global__ void up2_kernel(const float* __restrict__ in,
                           float* __restrict__ out, int Rin) {
    const int Rout = 2 * Rin - 1;
    const int idx = blockIdx.x * blockDim.x + threadIdx.x;
    if (idx >= Rout * Rout) return;
    const int i = idx / Rout, j = idx - i * Rout;
    const int i0 = i >> 1, j0 = j >> 1;
    const float* r0 = in + i0 * Rin;
    float v;
    if (!(i & 1) && !(j & 1))      v = r0[j0];
    else if (!(i & 1))             v = 0.5f * (r0[j0] + r0[j0 + 1]);
    else if (!(j & 1))             v = 0.5f * (r0[j0] + r0[j0 + Rin]);
    else v = 0.25f * (r0[j0] + r0[j0 + 1] + r0[j0 + Rin] + r0[j0 + Rin + 1]);
    out[idx] = v;
}

// ---------------------------------------------------------------------------
// boundary0 = (0 < up2(occ>0.5) < 1): true iff the contributing corner masks
// disagree (only possible at odd positions).
// ---------------------------------------------------------------------------
__global__ void boundary0_kernel(const float* __restrict__ occ,
                                 unsigned char* __restrict__ bnd, int Rin) {
    const int Rout = 2 * Rin - 1;
    const int idx = blockIdx.x * blockDim.x + threadIdx.x;
    if (idx >= Rout * Rout) return;
    const int i = idx / Rout, j = idx - i * Rout;
    const int i0 = i >> 1, j0 = j >> 1;
    const float* r0 = occ + i0 * Rin;
    const bool m00 = r0[j0] > 0.5f;
    bool res = false;
    if ((i & 1) && (j & 1)) {
        bool m01 = r0[j0 + 1] > 0.5f;
        bool m10 = r0[j0 + Rin] > 0.5f;
        bool m11 = r0[j0 + Rin + 1] > 0.5f;
        res = !((m00 == m01) && (m00 == m10) && (m00 == m11));
    } else if (j & 1) {
        res = (m00 != (r0[j0 + 1] > 0.5f));
    } else if (i & 1) {
        res = (m00 != (r0[j0 + Rin] > 0.5f));
    }
    bnd[idx] = (unsigned char)res;
}

__device__ __forceinline__ bool any3x3(const unsigned char* m, int i, int j, int R) {
    bool any = false;
#pragma unroll
    for (int di = -1; di <= 1; ++di) {
#pragma unroll
        for (int dj = -1; dj <= 1; ++dj) {
            int ii = i + di, jj = j + dj;
            if (ii >= 0 && ii < R && jj >= 0 && jj < R && m[ii * R + jj]) any = true;
        }
    }
    return any;
}

// ---------------------------------------------------------------------------
// Level merge: boundary = any3x3(boundary0) & ~calc; select q on boundary.
// ---------------------------------------------------------------------------
__global__ void combine_kernel(float* __restrict__ occ,
                               const float* __restrict__ q,
                               const unsigned char* __restrict__ bnd0,
                               const unsigned char* __restrict__ calc_prev,
                               unsigned char* __restrict__ conflicts,
                               unsigned char* __restrict__ calc, int R) {
    const int idx = blockIdx.x * blockDim.x + threadIdx.x;
    if (idx >= R * R) return;
    const int i = idx / R, j = idx - i * R;
    const int Rp = (R + 1) >> 1;
    const unsigned char ci = (!(i & 1) && !(j & 1))
                                 ? calc_prev[(i >> 1) * Rp + (j >> 1)] : 0;
    const bool b = any3x3(bnd0, i, j, R) && !ci;
    const float ov = occ[idx], qv = q[idx];
    const bool cf = b && ((ov - 0.5f) * (qv - 0.5f) < 0.f);
    if (b) occ[idx] = qv;
    conflicts[idx] = (unsigned char)cf;
    calc[idx] = (unsigned char)(ci | (unsigned char)b);
}

// ---------------------------------------------------------------------------
// One conflict-resolution sweep (double-buffered conflict masks).
// ---------------------------------------------------------------------------
__global__ void conflict_iter_kernel(float* __restrict__ occ,
                                     const float* __restrict__ q,
                                     const unsigned char* __restrict__ cin,
                                     unsigned char* __restrict__ cout,
                                     unsigned char* __restrict__ calc, int R) {
    const int idx = blockIdx.x * blockDim.x + threadIdx.x;
    if (idx >= R * R) return;
    const int i = idx / R, j = idx - i * R;
    const bool cand = any3x3(cin, i, j, R) && !calc[idx];
    const float ov = occ[idx], qv = q[idx];
    const bool cf = cand && ((ov - 0.5f) * (qv - 0.5f) < 0.f);
    if (cand) { occ[idx] = qv; calc[idx] = 1; }
    cout[idx] = (unsigned char)cf;
}

__global__ void fill_u8_kernel(unsigned char* __restrict__ p, int n,
                               unsigned char v) {
    int idx = blockIdx.x * blockDim.x + threadIdx.x;
    if (idx < n) p[idx] = v;
}

// ---------------------------------------------------------------------------
// Host-side launch sequence (graph-capture safe: stream-only, no allocs).
// ---------------------------------------------------------------------------
extern "C" void kernel_launch(void* const* d_in, const int* in_sizes, int n_in,
                              void* d_out, int out_size, void* d_ws, size_t ws_size,
                              hipStream_t stream) {
    (void)in_sizes; (void)n_in; (void)out_size; (void)ws_size;
    const float* W1   = (const float*)d_in[0];
    const float* b1   = (const float*)d_in[1];
    const float* W2   = (const float*)d_in[2];
    const float* b2   = (const float*)d_in[3];
    const float* W3   = (const float*)d_in[4];
    const float* b3   = (const float*)d_in[5];
    const float* bmin = (const float*)d_in[6];
    const float* bmax = (const float*)d_in[7];

    static const int RES[5] = {33, 65, 129, 257, 513};

    // Carve the workspace (deterministic offsets, 256B aligned).
    char* base = (char*)d_ws;
    size_t off = 0;
    auto carve = [&](size_t bytes) -> char* {
        char* p = base + off;
        off = (off + bytes + 255) & ~(size_t)255;
        return p;
    };
    _Float16* w2t = (_Float16*)carve(DHID * DHID * sizeof(_Float16));
    float* q[5];
    for (int l = 0; l < 5; ++l) q[l] = (float*)carve((size_t)RES[l] * RES[l] * 4);
    float* occbuf[5];
    occbuf[0] = q[0];                 // occ at level 0 is exactly q33
    for (int l = 1; l < 4; ++l)
        occbuf[l] = (float*)carve((size_t)RES[l] * RES[l] * 4);
    occbuf[4] = (float*)d_out;        // final 513^2 occ lives in d_out
    unsigned char* calc[5];
    for (int l = 0; l < 5; ++l)
        calc[l] = (unsigned char*)carve((size_t)RES[l] * RES[l]);
    unsigned char* cbufA = (unsigned char*)carve((size_t)513 * 513);
    unsigned char* cbufB = (unsigned char*)carve((size_t)513 * 513);

    // 1) W2 -> f16 transposed.
    w2_to_f16t_kernel<<<(DHID * DHID + 255) / 256, 256, 0, stream>>>(W2, w2t);

    // 2) Evaluate the MLP over every level's grid (WMMA).
    for (int l = 0; l < 5; ++l) {
        const int R = RES[l];
        const int npts = R * R;
        eval_grid_kernel<<<(npts + 127) / 128, 256, 0, stream>>>(
            W1, b1, b2, W3, b3, bmin, bmax, w2t, q[l], R);
    }

    // 3) Level 0: everything is computed.
    fill_u8_kernel<<<(RES[0] * RES[0] + 255) / 256, 256, 0, stream>>>(
        calc[0], RES[0] * RES[0], 1);

    // 4) Hierarchical refinement.
    float* occ_prev = occbuf[0];
    unsigned char* calc_prev = calc[0];
    for (int l = 1; l < 5; ++l) {
        const int R = RES[l], Rin = RES[l - 1];
        const int N = R * R;
        const int blk = (N + 255) / 256;
        float* occ = occbuf[l];

        up2_kernel<<<blk, 256, 0, stream>>>(occ_prev, occ, Rin);
        boundary0_kernel<<<blk, 256, 0, stream>>>(occ_prev, cbufA, Rin);
        combine_kernel<<<blk, 256, 0, stream>>>(occ, q[l], cbufA, calc_prev,
                                                cbufB, calc[l], R);
        conflict_iter_kernel<<<blk, 256, 0, stream>>>(occ, q[l], cbufB, cbufA,
                                                      calc[l], R);
        conflict_iter_kernel<<<blk, 256, 0, stream>>>(occ, q[l], cbufA, cbufB,
                                                      calc[l], R);
        conflict_iter_kernel<<<blk, 256, 0, stream>>>(occ, q[l], cbufB, cbufA,
                                                      calc[l], R);

        occ_prev = occ;
        calc_prev = calc[l];
    }
}
